// NearestSim_541165879321
// MI455X (gfx1250) — matrix operand order repaired and checked by reading, not compiled, not run on hardware
//
#include <hip/hip_runtime.h>
#include <hip/hip_bf16.h>

typedef __attribute__((ext_vector_type(16))) __bf16 bf16x16;
typedef __attribute__((ext_vector_type(8)))  __bf16 bf16x8;
typedef __attribute__((ext_vector_type(8)))  float  f32x8;

#define C_DIM           512
#define M_ITEMS         4096
#define T_ROWS          16384
#define ROWS_PER_BLOCK  64
#define WAVES_PER_BLOCK 4
#define THREADS         (WAVES_PER_BLOCK * 32)
#define KCHUNKS         (C_DIM / 32)
#define ITEM_TILES      (M_ITEMS / 16)
#define NB              4                       // item tiles per sweep (indep. acc chains)
#define TILE_GROUPS     (ITEM_TILES / NB)       // 64
#define ROW_STRIDE      (2 * C_DIM)             // interleaved hi|lo row, in bf16 elems

#define SHUF16(a, b) __builtin_shufflevector(a, b, 0,1,2,3,4,5,6,7,8,9,10,11,12,13,14,15)
#define WMMA_BF16(A, B, C) \
    __builtin_amdgcn_wmma_f32_16x16x32_bf16(false, A, false, B, (short)0, C, false, false)

// ---------------------------------------------------------------------------
// Pass 1: split items (f32) into interleaved bf16 hi/lo rows in workspace:
// pk[item][0..511] = hi, pk[item][512..1023] = lo.  x = hi + lo, err ~2^-16.
// ---------------------------------------------------------------------------
__global__ __launch_bounds__(256) void cvt_items_kernel(
    const float* __restrict__ items,
    unsigned short* __restrict__ pk_u16)
{
    __bf16* pk = (__bf16*)pk_u16;
    size_t i = (size_t)blockIdx.x * blockDim.x + threadIdx.x;   // exact grid
    size_t item = i >> 9;
    size_t c    = i & (C_DIM - 1);
    float x = items[i];
    __bf16 h = (__bf16)x;
    pk[item * ROW_STRIDE + c]         = h;
    pk[item * ROW_STRIDE + C_DIM + c] = (__bf16)(x - (float)h);
}

// ---------------------------------------------------------------------------
// Pass 2: per 16-row query tile (one wave), argmax over all items via
// bf16x3 WMMA GEMM (4 independent accumulator chains), then exact f32
// cosine against the winning item.
// ---------------------------------------------------------------------------
__global__ __launch_bounds__(THREADS) void nearest_sim_kernel(
    const float* __restrict__ q,                 // (16384, 512)
    const float* __restrict__ items,             // (4096, 512)
    const unsigned short* __restrict__ itpk_u16, // interleaved bf16 hi|lo (4096, 1024)
    float* __restrict__ out)                     // (16384,)
{
    // Interleaved hi|lo per row: all A accesses are base + small immediate.
    __shared__ __bf16 sa[ROWS_PER_BLOCK][ROW_STRIDE];   // 128 KB
    __shared__ int    s_idx[WAVES_PER_BLOCK][16];

    const __bf16* it_pk = (const __bf16*)itpk_u16;
    const int row0 = blockIdx.x * ROWS_PER_BLOCK;

    // Stage this block's query rows into LDS as interleaved bf16 hi/lo.
    for (int e = threadIdx.x; e < ROWS_PER_BLOCK * C_DIM; e += THREADS) {
        int r = e >> 9;               // / 512
        int c = e & (C_DIM - 1);
        float x = q[(size_t)(row0 + r) * C_DIM + c];
        __bf16 h = (__bf16)x;
        sa[r][c]         = h;
        sa[r][C_DIM + c] = (__bf16)(x - (float)h);
    }
    __syncthreads();

    const int wave   = threadIdx.x >> 5;
    const int lane   = threadIdx.x & 31;
    const int laneM  = lane & 15;     // A row / B col / D col within tile
    const int laneHi = lane >> 4;     // selects K sub-range per ISA layout

    // A lane base: row (wave*16 + laneM), K-subrange by lane half {0..7,16..23}/{8..15,24..31}
    const __bf16* ap = &sa[wave * 16 + laneM][laneHi * 8];

    float best[8];
    int   bidx[8];
#pragma unroll
    for (int r = 0; r < 8; ++r) { best[r] = -3.4e38f; bidx[r] = 0; }

    for (int g = 0; g < TILE_GROUPS; ++g) {
        f32x8 acc0 = {}, acc1 = {}, acc2 = {}, acc3 = {};
        // B lane base: item row (g*64 + laneM); K-subrange by lane half {0..7,16..23}/{8..15,24..31}.
        // Tile (+t*16 rows -> +32KB) and hi/lo plane (+1KB) are immediates.
        const __bf16* bp = it_pk + (size_t)(g * (NB * 16) + laneM) * ROW_STRIDE
                                 + laneHi * 8;

#pragma unroll 2
        for (int kc = 0; kc < KCHUNKS; ++kc) {
            bf16x8 ah0 = *(const bf16x8*)(ap + kc * 32);
            bf16x8 ah1 = *(const bf16x8*)(ap + kc * 32 + 16);
            bf16x8 al0 = *(const bf16x8*)(ap + C_DIM + kc * 32);
            bf16x8 al1 = *(const bf16x8*)(ap + C_DIM + kc * 32 + 16);
            bf16x16 Ahi = SHUF16(ah0, ah1);
            bf16x16 Alo = SHUF16(al0, al1);

#define DO_TILE(t, ACC)                                                              \
            {                                                                        \
                const __bf16* tb = bp + (t) * 16 * ROW_STRIDE + kc * 32;             \
                bf16x8 bh0 = *(const bf16x8*)(tb);                                   \
                bf16x8 bh1 = *(const bf16x8*)(tb + 16);                              \
                bf16x8 bl0 = *(const bf16x8*)(tb + C_DIM);                           \
                bf16x8 bl1 = *(const bf16x8*)(tb + C_DIM + 16);                      \
                bf16x16 Bhi = SHUF16(bh0, bh1);                                      \
                bf16x16 Blo = SHUF16(bl0, bl1);                                      \
                ACC = WMMA_BF16(Ahi, Bhi, ACC);                                      \
                ACC = WMMA_BF16(Ahi, Blo, ACC);                                      \
                ACC = WMMA_BF16(Alo, Bhi, ACC);                                      \
            }
            DO_TILE(0, acc0)
            DO_TILE(1, acc1)
            DO_TILE(2, acc2)
            DO_TILE(3, acc3)
#undef DO_TILE
        }

        // D layout: VGPR r, lane L -> row (r + 8*(L>>4)), col (L&15).
        int colBase = g * (NB * 16) + laneM;
#define DO_ARGMAX(ACC, COL)                                                          \
        {                                                                            \
            _Pragma("unroll")                                                        \
            for (int r = 0; r < 8; ++r) {                                            \
                if ((ACC)[r] > best[r]) { best[r] = (ACC)[r]; bidx[r] = (COL); }     \
            }                                                                        \
        }
        DO_ARGMAX(acc0, colBase +  0)
        DO_ARGMAX(acc1, colBase + 16)
        DO_ARGMAX(acc2, colBase + 32)
        DO_ARGMAX(acc3, colBase + 48)
#undef DO_ARGMAX
    }

    // Argmax reduction across the 16 lanes of each half (rows r and r+8).
#pragma unroll
    for (int r = 0; r < 8; ++r) {
        float v  = best[r];
        int   ix = bidx[r];
#pragma unroll
        for (int d = 1; d < 16; d <<= 1) {
            float ov = __shfl_xor(v, d, 32);
            int   oi = __shfl_xor(ix, d, 32);
            if (ov > v || (ov == v && oi < ix)) { v = ov; ix = oi; }  // top_k tie: lower index
        }
        if (lane == 0)  s_idx[wave][r]     = ix;
        if (lane == 16) s_idx[wave][r + 8] = ix;
    }
    __syncthreads();

    // Exact f32 negative cosine similarity for each row's winning item.
    for (int m = 0; m < 16; ++m) {
        int grow = row0 + wave * 16 + m;
        int bi   = s_idx[wave][m];
        const float* qp = q     + (size_t)grow * C_DIM;
        const float* ip = items + (size_t)bi   * C_DIM;
        float dot = 0.f, q2 = 0.f, i2 = 0.f;
#pragma unroll 4
        for (int t = lane; t < C_DIM; t += 32) {
            float a = qp[t], b = ip[t];
            dot += a * b; q2 += a * a; i2 += b * b;
        }
#pragma unroll
        for (int d = 1; d < 32; d <<= 1) {
            dot += __shfl_xor(dot, d, 32);
            q2  += __shfl_xor(q2,  d, 32);
            i2  += __shfl_xor(i2,  d, 32);
        }
        if (lane == 0) {
            float nq = fmaxf(sqrtf(q2), 1e-12f);
            float ni = fmaxf(sqrtf(i2), 1e-12f);
            out[grow] = -dot / (nq * ni);
        }
    }
}

extern "C" void kernel_launch(void* const* d_in, const int* in_sizes, int n_in,
                              void* d_out, int out_size, void* d_ws, size_t ws_size,
                              hipStream_t stream) {
    const float* q     = (const float*)d_in[0];   // (8, 2048, 512) f32
    const float* items = (const float*)d_in[1];   // (4096, 512)   f32
    float* out = (float*)d_out;                   // (8, 2048)     f32

    unsigned short* it_pk = (unsigned short*)d_ws;    // 8 MB interleaved hi|lo planes

    cvt_items_kernel<<<(M_ITEMS * C_DIM) / 256, 256, 0, stream>>>(items, it_pk);
    nearest_sim_kernel<<<T_ROWS / ROWS_PER_BLOCK, THREADS, 0, stream>>>(
        q, items, it_pk, out);
}
